// RVQTokenizer_20813411516941
// MI455X (gfx1250) — matrix operand order repaired and checked by reading, not compiled, not run on hardware
//
#include <hip/hip_runtime.h>
#include <hip/hip_bf16.h>

// ---------------------------------------------------------------------------
// RVQ tokenizer for MI455X (gfx1250, wave32, WMMA bf16 16x16x32)
//   kernel 1: fused conv1(1->256,k3) + relu + conv2(256->128,k3) + relu + mean_t
//             one workgroup (256 thr = 8 waves) per batch row; GEMM M=128(l),
//             N=time, K=768=(k,c) via v_wmma_f32_16x16x32_bf16 from LDS.
//   kernel 2: 12-stage residual VQ, scores via WMMA (16 rows x 512 codes),
//             argmin + residual update per stage.
// ---------------------------------------------------------------------------

typedef __attribute__((ext_vector_type(16))) __bf16 v16bf;
typedef __attribute__((ext_vector_type(8)))  __bf16 v8bf;
typedef __attribute__((ext_vector_type(8)))  float  v8f;
typedef __attribute__((ext_vector_type(4)))  float  v4f;

#define FEAT 840
#define HID  256
#define LAT  128
#define NQV  12
#define NEMB 512
#define K2   768                 // HID * 3 (im2col contraction)
#define ASTR 776                 // a_lds row stride (bf16), padded: 388 dw % 64 = 4
#define HSTR 264                 // h1s col stride (bf16), padded: 132 dw % 64 = 4
#define NTC  128                 // time-chunk
#define NCHK 7                   // ceil(840/128)

// dynamic LDS partition (bytes)
#define SM_A  0                          // w2 repacked bf16 [128][776]  = 198656 B
#define SM_H  198656                     // h1 chunk bf16 [132 cols][264] = 69696 B
#define SM_X  268352                     // x row f32 [844]               = 3376 B
#define SM_Z  271728                     // z partial f32 [128]           = 512 B
#define SM_SZ 272240

__device__ __forceinline__ v16bf frag16(const __bf16* p0, const __bf16* p1) {
  union { v16bf v; v8bf h[2]; } u;
  u.h[0] = *(const v8bf*)p0;     // ds_read_b128
  u.h[1] = *(const v8bf*)p1;     // ds_read_b128
  return u.v;
}
__device__ __forceinline__ v8f wmma_bf16(v16bf a, v16bf b, v8f c) {
  return __builtin_amdgcn_wmma_f32_16x16x32_bf16(false, a, false, b, (short)0, c,
                                                 false, false);
}

__global__ __launch_bounds__(256)
void encode_fused_kernel(const float* __restrict__ x,  const float* __restrict__ w1,
                         const float* __restrict__ b1, const float* __restrict__ w2,
                         const float* __restrict__ b2, float* __restrict__ zout) {
  extern __shared__ char smem[];
  __bf16* a_lds = (__bf16*)(smem + SM_A);   // A[l][kk], kk = k*256 + c
  __bf16* h1s   = (__bf16*)(smem + SM_H);   // [col][c], col = t - t0 + 1
  float*  xs    = (float*) (smem + SM_X);   // x row, zero padded by 1 each side
  float*  zsum  = (float*) (smem + SM_Z);

  const int tid = threadIdx.x;
  const int bb  = blockIdx.x;

  // ---- stage x row into LDS (padded) ----
  for (int i = tid; i < FEAT + 2; i += 256) {
    float v = 0.f;
    if (i >= 1 && i <= FEAT) v = x[(size_t)bb * FEAT + (i - 1)];
    xs[i] = v;
  }
  if (tid < LAT) zsum[tid] = 0.f;

  // ---- repack w2 -> bf16 A matrix in LDS: a[l][k*256+c] = w2[l][c][k] ----
  for (int i = tid; i < LAT * K2; i += 256) {
    int l  = i / K2;
    int kk = i - l * K2;
    int c  = kk & 255, k = kk >> 8;
    a_lds[l * ASTR + kk] = (__bf16)w2[l * K2 + c * 3 + k];
  }

  // per-thread conv1 channel
  const int   cch = tid;
  const float w10 = w1[cch * 3 + 0], w11 = w1[cch * 3 + 1], w12 = w1[cch * 3 + 2];
  const float b1c = b1[cch];

  const int wave = tid >> 5;           // M tile: l in [16*wave, 16*wave+16)
  const int lane = tid & 31;
  const int ln   = lane & 15;          // column-in-tile (N) / row-in-tile (A)
  const int g    = lane >> 4;

  float b2r[8];
#pragma unroll
  for (int j = 0; j < 8; ++j) b2r[j] = b2[wave * 16 + 8 * g + j];  // C row m = 8g+j

  v8f zacc = {};                       // masked relu accumulation over t

  __syncthreads();

  for (int chunk = 0; chunk < NCHK; ++chunk) {
    const int t0 = chunk * NTC;
    // ---- conv1 for this chunk: h1s[col][c], col 0..129 <-> t = t0+col-1 ----
    for (int col = 0; col < NTC + 2; ++col) {
      int   t = t0 + col - 1;
      float v = 0.f;
      if (t >= 0 && t < FEAT) {
        float a = b1c + xs[t] * w10 + xs[t + 1] * w11 + xs[t + 2] * w12;
        v = a > 0.f ? a : 0.f;
      }
      h1s[col * HSTR + cch] = (__bf16)v;
    }
    __syncthreads();

    // ---- WMMA GEMM over this chunk: pairs of N tiles per wave ----
    const __bf16* arow = a_lds + (wave * 16 + ln) * ASTR;
    for (int nt = 0; nt < 8; nt += 2) {
      v8f c0 = {}, c1 = {};
#pragma unroll
      for (int ks = 0; ks < 24; ++ks) {
        const int k  = ks >> 3;             // conv tap 0..2
        const int cb = (ks & 7) * 32;       // channel base
        const __bf16* ap = arow + ks * 32 + g * 16;
        v16bf A  = frag16(ap, ap + 8);
        const __bf16* bp0 = h1s + (nt * 16 + ln + k) * HSTR + cb + g * 16;
        const __bf16* bp1 = bp0 + 16 * HSTR;
        v16bf B0 = frag16(bp0, bp0 + 8);
        v16bf B1 = frag16(bp1, bp1 + 8);
        c0 = wmma_bf16(A, B0, c0);
        c1 = wmma_bf16(A, B1, c1);
      }
      // epilogue: bias + relu + time-mask, accumulate (rows fixed per lane)
      const int   tc0 = t0 + nt * 16 + ln;
      const int   tc1 = tc0 + 16;
      const float s0  = (tc0 < FEAT) ? 1.f : 0.f;
      const float s1  = (tc1 < FEAT) ? 1.f : 0.f;
#pragma unroll
      for (int j = 0; j < 8; ++j) {
        float v0 = c0[j] + b2r[j]; v0 = v0 > 0.f ? v0 : 0.f;
        float v1 = c1[j] + b2r[j]; v1 = v1 > 0.f ? v1 : 0.f;
        zacc[j] += v0 * s0 + v1 * s1;
      }
    }
    __syncthreads();   // before next chunk overwrites h1s
  }

  // ---- reduce per-lane partials across the 16 N-classes (ds_add_f32) ----
#pragma unroll
  for (int j = 0; j < 8; ++j) atomicAdd(&zsum[wave * 16 + 8 * g + j], zacc[j]);
  __syncthreads();
  if (tid < LAT) zout[(size_t)bb * LAT + tid] = zsum[tid] * (1.0f / (float)FEAT);
}

// ---------------------------------------------------------------------------
// kernel 2: residual VQ, one wave per 16 batch rows
// ---------------------------------------------------------------------------
__global__ __launch_bounds__(32)
void rvq_kernel(const float* __restrict__ cbk, float* __restrict__ zio,
                float* __restrict__ idx_out) {
  __shared__ float r[16][132];       // residual (padded vs LDS banks)
  __shared__ float zq[16][132];
  __shared__ float cnorm[NEMB];
  __shared__ float redv[16][16];
  __shared__ int   redi[16][16];
  __shared__ int   sidx[16];

  const int lane = threadIdx.x;
  const int b0   = blockIdx.x * 16;
  const int ln   = lane & 15, g = lane >> 4;

  for (int i = lane; i < 16 * LAT; i += 32) {
    int m = i >> 7, kk = i & 127;
    r[m][kk]  = zio[(size_t)(b0 + m) * LAT + kk];
    zq[m][kk] = 0.f;
  }
  __syncthreads();

  for (int q = 0; q < NQV; ++q) {
    const float* cb = cbk + (size_t)q * NEMB * LAT;
    // code norms
    for (int n = lane; n < NEMB; n += 32) {
      const v4f* p = (const v4f*)(cb + (size_t)n * LAT);
      float s = 0.f;
#pragma unroll 8
      for (int i = 0; i < LAT / 4; ++i) {
        v4f f = p[i];
        s += f[0] * f[0] + f[1] * f[1] + f[2] * f[2] + f[3] * f[3];
      }
      cnorm[n] = s;
    }
    __syncthreads();

    // residual -> bf16 A fragments (row m = ln)
    v16bf Afr[4];
#pragma unroll
    for (int ks = 0; ks < 4; ++ks) {
      union { v16bf v; __bf16 e[16]; } u;
      const int base0 = ks * 32 + g * 16;
#pragma unroll
      for (int i = 0; i < 8; ++i) {
        u.e[i]     = (__bf16)r[ln][base0 + i];
        u.e[i + 8] = (__bf16)r[ln][base0 + 8 + i];
      }
      Afr[ks] = u.v;
    }

    float mval[8]; int midx[8];
#pragma unroll
    for (int j = 0; j < 8; ++j) { mval[j] = 3.0e38f; midx[j] = 0; }

    for (int nt = 0; nt < NEMB / 16; ++nt) {
      v8f c = {};
      const int ng = nt * 16 + ln;        // this lane's code column
#pragma unroll
      for (int ks = 0; ks < 4; ++ks) {
        const v4f* bp = (const v4f*)(cb + (size_t)ng * LAT + ks * 32 + g * 16);
        union { v16bf v; __bf16 e[16]; } ub;
#pragma unroll
        for (int i = 0; i < 4; ++i) {
          v4f f = bp[i];
          ub.e[4 * i + 0] = (__bf16)f[0];
          ub.e[4 * i + 1] = (__bf16)f[1];
          ub.e[4 * i + 2] = (__bf16)f[2];
          ub.e[4 * i + 3] = (__bf16)f[3];
        }
        c = wmma_bf16(Afr[ks], ub.v, c);
      }
      const float cn = cnorm[ng];
#pragma unroll
      for (int j = 0; j < 8; ++j) {
        float d = cn - 2.f * c[j];
        if (d < mval[j]) { mval[j] = d; midx[j] = ng; }
      }
    }

    // cross-lane argmin per row m = 8g + j
#pragma unroll
    for (int j = 0; j < 8; ++j) { redv[8 * g + j][ln] = mval[j]; redi[8 * g + j][ln] = midx[j]; }
    __syncthreads();
    if (lane < 16) {
      float bv = redv[lane][0]; int bi = redi[lane][0];
      for (int t = 1; t < 16; ++t)
        if (redv[lane][t] < bv) { bv = redv[lane][t]; bi = redi[lane][t]; }
      sidx[lane] = bi;
      idx_out[(size_t)(b0 + lane) * NQV + q] = (float)bi;
    }
    __syncthreads();

    // residual update + zq accumulation
    for (int i = lane; i < 16 * LAT; i += 32) {
      int m = i >> 7, kk = i & 127;
      float qv = cb[(size_t)sidx[m] * LAT + kk];
      r[m][kk]  -= qv;
      zq[m][kk] += qv;
    }
    __syncthreads();
  }

  for (int i = lane; i < 16 * LAT; i += 32) {
    int m = i >> 7, kk = i & 127;
    zio[(size_t)(b0 + m) * LAT + kk] = zq[m][kk];
  }
}

extern "C" void kernel_launch(void* const* d_in, const int* in_sizes, int n_in,
                              void* d_out, int out_size, void* d_ws, size_t ws_size,
                              hipStream_t stream) {
  const float* x   = (const float*)d_in[0];
  const float* w1  = (const float*)d_in[1];
  const float* b1  = (const float*)d_in[2];
  const float* w2  = (const float*)d_in[3];
  const float* b2  = (const float*)d_in[4];
  const float* cbk = (const float*)d_in[5];
  float* out = (float*)d_out;                    // [0,2048*128): zq (z staged here)
  float* idx = out + (size_t)2048 * LAT;         // [.., +2048*12): indices as float

  encode_fused_kernel<<<2048, 256, SM_SZ, stream>>>(x, w1, b1, w2, b2, out);
  rvq_kernel<<<2048 / 16, 32, 0, stream>>>(cbk, out, idx);
}
